// RNNCellStacked_51934744543239
// MI455X (gfx1250) — matrix operand order repaired and checked by reading, not compiled, not run on hardware
//
#include <hip/hip_runtime.h>
#include <hip/hip_bf16.h>
#include <math.h>

// ---------------------------------------------------------------------------
// Stacked RNN (2 cells), B=256, H=1024, T=64, fp32 reference.
//   - bf16 WMMA (v_wmma_f32_16x16x32_bf16), f32 accumulation.
//   - Fused dual GEMM via K-concat: A=[x|h] (K=2048) vs Wc=[W_ih|W_hh].
//   - ONE persistent kernel for the whole 128-phase recurrence; phases
//     separated by a device-wide atomic barrier (no per-step launch latency).
//   - Time loop unrolled by 2 so buffer parity is compile-time constant:
//     every activation pointer is a constant-offset GEP of a kernel arg,
//     keeping all matrix loads on the GLOBAL path (no FLAT fallback).
//   - 128 persistent workgroups (grid 8x16), block tile 32x64,
//     wave tile 16x32 (2 WMMAs / wave / k-step).
//   - Deterministic 2-stage loss reduction.
// ---------------------------------------------------------------------------

typedef __bf16 bf16_t;
typedef __attribute__((ext_vector_type(8)))  bf16_t v8bf;
typedef __attribute__((ext_vector_type(16))) bf16_t v16bf;
typedef __attribute__((ext_vector_type(8)))  float  v8f;

#define RNN_HID    1024
#define RNN_K2     2048      // concat K = 2*HID
#define RNN_BATCH  256
#define RNN_T      64
#define RNN_LAYERS 2

#define GRID_M     8         // 256 / 32
#define GRID_N     16        // 1024 / 64
#define NBLOCKS    (GRID_M * GRID_N)   // 128 persistent workgroups

#define ACT_ELEMS  (RNN_BATCH * RNN_K2)   // elements per activation buffer

// ---------------- workspace layout (bytes) ----------------
#define WS_WC_OFF     0                                          // bf16 [2][1024][2048] = 8 MB
#define WS_WC_BYTES   (RNN_LAYERS * RNN_HID * RNN_K2 * 2)
#define WS_BIAS_OFF   (WS_WC_OFF + WS_WC_BYTES)                  // f32 [2][1024]
#define WS_BIAS_BYTES (RNN_LAYERS * RNN_HID * 4)
#define WS_ACT_OFF    (WS_BIAS_OFF + WS_BIAS_BYTES)              // bf16 [4][B][K2]
#define WS_ACT_BYTES  (ACT_ELEMS * 2)                            // 1 MB per buffer
#define WS_PART_OFF   (WS_ACT_OFF + 4 * WS_ACT_BYTES)            // f32 [1024] partials
#define WS_CTR_OFF    (WS_PART_OFF + 1024 * 4)                   // u32 barrier counter

#define LOSS_BLOCKS  1024
#define OUT_ELEMS    (RNN_BATCH * RNN_T * RNN_HID)

// Activation buffer indices within the act base region:
//   [0] act0 parity0   [1] act0 parity1   [2] act1 parity0   [3] act1 parity1
#define A0(p) ((size_t)(p) * ACT_ELEMS)
#define A1(p) ((size_t)(2 + (p)) * ACT_ELEMS)

// ---------------------------------------------------------------------------
// Prep kernels
// ---------------------------------------------------------------------------

// Wc[layer][j][k] = bf16( k<H ? w_ih[layer][j][k] : w_hh[layer][j][k-H] )
__global__ __launch_bounds__(256) void prep_weights_kernel(
    const float* __restrict__ w_ih, const float* __restrict__ w_hh,
    bf16_t* __restrict__ Wc)
{
    const int total = RNN_LAYERS * RNN_HID * RNN_K2;
    for (int idx = blockIdx.x * blockDim.x + threadIdx.x; idx < total;
         idx += gridDim.x * blockDim.x) {
        int layer = idx / (RNN_HID * RNN_K2);
        int rem   = idx - layer * (RNN_HID * RNN_K2);
        int j     = rem / RNN_K2;
        int k     = rem - j * RNN_K2;
        float w;
        if (k < RNN_HID)
            w = w_ih[(layer * RNN_HID + j) * RNN_HID + k];
        else
            w = w_hh[(layer * RNN_HID + j) * RNN_HID + (k - RNN_HID)];
        Wc[idx] = (bf16_t)w;
    }
}

// bias[layer][j] = b_ih + b_hh (f32); thread 0 also resets the grid barrier
__global__ __launch_bounds__(256) void prep_bias_kernel(
    const float* __restrict__ b_ih, const float* __restrict__ b_hh,
    float* __restrict__ bias, unsigned int* __restrict__ ctr)
{
    int idx = blockIdx.x * blockDim.x + threadIdx.x;
    if (idx < RNN_LAYERS * RNN_HID) bias[idx] = b_ih[idx] + b_hh[idx];
    if (idx == 0) *ctr = 0u;
}

// act[A0(0)] = [bf16(x0) | 0] ; act[A1(0)] = 0
__global__ __launch_bounds__(256) void prep_state_kernel(
    const float* __restrict__ x0, bf16_t* __restrict__ act)
{
    const int total = ACT_ELEMS;
    for (int idx = blockIdx.x * blockDim.x + threadIdx.x; idx < total;
         idx += gridDim.x * blockDim.x) {
        int b = idx / RNN_K2;
        int k = idx - b * RNN_K2;
        act[A0(0) + idx] = (k < RNN_HID) ? (bf16_t)x0[b * RNN_HID + k]
                                         : (bf16_t)0.0f;
        act[A1(0) + idx] = (bf16_t)0.0f;
    }
}

// ---------------------------------------------------------------------------
// Device-wide barrier: monotonic counter, release-arrive / acquire-spin.
// ---------------------------------------------------------------------------
__device__ __forceinline__ void grid_barrier(unsigned int* ctr, int phase)
{
    __syncthreads();
    if (threadIdx.x == 0) {
        __hip_atomic_fetch_add(ctr, 1u, __ATOMIC_RELEASE,
                               __HIP_MEMORY_SCOPE_AGENT);
        const unsigned int target = (unsigned int)NBLOCKS * (unsigned int)phase;
        while (__hip_atomic_load(ctr, __ATOMIC_ACQUIRE,
                                 __HIP_MEMORY_SCOPE_AGENT) < target) {
            __builtin_amdgcn_s_sleep(1);
        }
    }
    __syncthreads();
}

// ---------------------------------------------------------------------------
// One RNN-cell tile: D[16x32 per wave] = tanh(A[16,2048] @ W^T + bias).
// Wave tile 16(M) x 32(N): one A fragment feeds two WMMAs per k-step.
// ---------------------------------------------------------------------------
__device__ __forceinline__ void cell_gemm_tile(
    const bf16_t* __restrict__ A,     // [BATCH, K2]
    const bf16_t* __restrict__ W,     // [HID,  K2]
    const float*  __restrict__ bias,  // [HID]
    bf16_t* __restrict__ dst0, int c0,
    bf16_t* __restrict__ dst1, int c1,
    float*  __restrict__ outf, int outf_stride,
    int m_wave, int n_wave, int lane)
{
    const int half = lane >> 4;    // 0: lanes 0-15, 1: lanes 16-31
    const int l16  = lane & 15;

    v8f acc0 = {}, acc1 = {};

    const bf16_t* a_row = A + (size_t)(m_wave + l16) * RNN_K2;
    const bf16_t* b_c0  = W + (size_t)(n_wave + l16) * RNN_K2;
    const bf16_t* b_c1  = W + (size_t)(n_wave + 16 + l16) * RNN_K2;

    const int a_off = half * 16;   // A 16x32: lanes 0-15 K=0..15, 16-31 K=16..31
    const int b_off = half * 16;   // B 32x16: lanes 0-15 K=0..15, 16-31 K=16..31

    #pragma unroll 4
    for (int kb = 0; kb < RNN_K2; kb += 32) {
        v8bf alo = *(const v8bf*)(a_row + kb + a_off);
        v8bf ahi = *(const v8bf*)(a_row + kb + a_off + 8);
        v16bf a = __builtin_shufflevector(alo, ahi,
                    0,1,2,3,4,5,6,7,8,9,10,11,12,13,14,15);
        v16bf b0 = *(const v16bf*)(b_c0 + kb + b_off);
        v16bf b1 = *(const v16bf*)(b_c1 + kb + b_off);

        acc0 = __builtin_amdgcn_wmma_f32_16x16x32_bf16(
                   false, a, false, b0, (short)0, acc0, false, false);
        acc1 = __builtin_amdgcn_wmma_f32_16x16x32_bf16(
                   false, a, false, b1, (short)0, acc1, false, false);
    }

    // Epilogue: C/D layout — lane column = l16, row = m_wave + half*8 + v.
    v8f accs[2] = {acc0, acc1};
    #pragma unroll
    for (int ni = 0; ni < 2; ++ni) {
        const int n  = n_wave + ni * 16 + l16;
        const float bn = bias[n];
        #pragma unroll
        for (int v = 0; v < 8; ++v) {
            const int m = m_wave + half * 8 + v;
            float h = tanhf(accs[ni][v] + bn);
            bf16_t hb = (bf16_t)h;
            dst0[(size_t)m * RNN_K2 + c0 + n] = hb;
            dst1[(size_t)m * RNN_K2 + c1 + n] = hb;
            if (outf) outf[(size_t)m * outf_stride + n] = h;
        }
    }
}

// One (step, layer-pair) with compile-time parity P.
template <int P>
__device__ __forceinline__ void rnn_step_pair(
    bf16_t* __restrict__ act,         // base of 4 activation buffers
    const bf16_t* __restrict__ Wc, const float* __restrict__ bias,
    float* __restrict__ out_t,        // ret[:, t, :] base (= out + t*HID)
    unsigned int* __restrict__ ctr, int& phase,
    int m_wave, int n_wave, int lane)
{
    // layer 0: h0 -> act1[P][:,0:H] (layer-1 input this step)
    //              + act0[1-P][:,H:2H] (h0_prev for next step)
    cell_gemm_tile(act + A0(P), Wc, bias,
                   act + A1(P), 0,
                   act + A0(1 - P), RNN_HID,
                   (float*)nullptr, 0,
                   m_wave, n_wave, lane);
    grid_barrier(ctr, ++phase);
    // layer 1: h1 -> act0[1-P][:,0:H] (layer-0 input next step)
    //              + act1[1-P][:,H:2H] (h1_prev) + f32 ret[:,t,:]
    cell_gemm_tile(act + A1(P), Wc + (size_t)RNN_HID * RNN_K2, bias + RNN_HID,
                   act + A0(1 - P), 0,
                   act + A1(1 - P), RNN_HID,
                   out_t, RNN_T * RNN_HID,
                   m_wave, n_wave, lane);
    grid_barrier(ctr, ++phase);
}

// ---------------------------------------------------------------------------
// Persistent recurrence kernel: 64 steps x 2 layers in one launch.
// Grid (8,16) = 128 blocks x 128 threads; block tile 32(M) x 64(N).
// ---------------------------------------------------------------------------
__global__ __launch_bounds__(128) void rnn_persistent_kernel(
    const bf16_t* __restrict__ Wc,    // [2][HID][K2]
    const float*  __restrict__ bias,  // [2][HID]
    bf16_t* __restrict__ act,         // [4][BATCH][K2]
    float* __restrict__ out,          // ret [B][T][H]
    unsigned int* __restrict__ ctr)
{
    const int lane = threadIdx.x & 31;
    const int wave = threadIdx.x >> 5;
    const int m_wave = blockIdx.x * 32 + (wave >> 1) * 16;
    const int n_wave = blockIdx.y * 64 + (wave & 1) * 32;

    int phase = 0;
    for (int t = 0; t < RNN_T; t += 2) {
        rnn_step_pair<0>(act, Wc, bias, out + (size_t)t * RNN_HID,
                         ctr, phase, m_wave, n_wave, lane);
        rnn_step_pair<1>(act, Wc, bias, out + (size_t)(t + 1) * RNN_HID,
                         ctr, phase, m_wave, n_wave, lane);
    }
}

// ---------------------------------------------------------------------------
// Loss: mean((ret - labels)^2), deterministic two-stage reduction
// ---------------------------------------------------------------------------
__global__ __launch_bounds__(256) void loss_partial_kernel(
    const float* __restrict__ ret, const float* __restrict__ labels,
    float* __restrict__ partial)
{
    __shared__ float sm[256];
    float s = 0.0f;
    for (int i = blockIdx.x * blockDim.x + threadIdx.x; i < OUT_ELEMS;
         i += gridDim.x * blockDim.x) {
        float d = ret[i] - labels[i];
        s += d * d;
    }
    sm[threadIdx.x] = s;
    __syncthreads();
    for (int off = 128; off > 0; off >>= 1) {
        if ((int)threadIdx.x < off) sm[threadIdx.x] += sm[threadIdx.x + off];
        __syncthreads();
    }
    if (threadIdx.x == 0) partial[blockIdx.x] = sm[0];
}

__global__ __launch_bounds__(256) void loss_final_kernel(
    const float* __restrict__ partial, float* __restrict__ out_loss)
{
    __shared__ float sm[256];
    float s = 0.0f;
    for (int i = threadIdx.x; i < LOSS_BLOCKS; i += 256) s += partial[i];
    sm[threadIdx.x] = s;
    __syncthreads();
    for (int off = 128; off > 0; off >>= 1) {
        if ((int)threadIdx.x < off) sm[threadIdx.x] += sm[threadIdx.x + off];
        __syncthreads();
    }
    if (threadIdx.x == 0) *out_loss = sm[0] * (1.0f / (float)OUT_ELEMS);
}

// ---------------------------------------------------------------------------
// Launch
// ---------------------------------------------------------------------------
extern "C" void kernel_launch(void* const* d_in, const int* in_sizes, int n_in,
                              void* d_out, int out_size, void* d_ws, size_t ws_size,
                              hipStream_t stream) {
    const float* x0     = (const float*)d_in[0];  // [1,256,1024]
    const float* labels = (const float*)d_in[1];  // [1,256,64,1024]
    const float* w_ih   = (const float*)d_in[2];  // [2,1024,1024]
    const float* w_hh   = (const float*)d_in[3];  // [2,1024,1024]
    const float* b_ih   = (const float*)d_in[4];  // [2,1024]
    const float* b_hh   = (const float*)d_in[5];  // [2,1024]

    float* out = (float*)d_out;                   // ret [256,64,1024] ++ loss[1]

    char* ws = (char*)d_ws;
    bf16_t* Wc      = (bf16_t*)(ws + WS_WC_OFF);
    float*  bias    = (float*)(ws + WS_BIAS_OFF);
    bf16_t* act     = (bf16_t*)(ws + WS_ACT_OFF);
    float*  partial = (float*)(ws + WS_PART_OFF);
    unsigned int* ctr = (unsigned int*)(ws + WS_CTR_OFF);

    // 1) prep (parallel, once per call; also resets the grid barrier)
    prep_weights_kernel<<<4096, 256, 0, stream>>>(w_ih, w_hh, Wc);
    prep_bias_kernel<<<(RNN_LAYERS * RNN_HID + 255) / 256, 256, 0, stream>>>(
        b_ih, b_hh, bias, ctr);
    prep_state_kernel<<<2048, 256, 0, stream>>>(x0, act);

    // 2) entire recurrence in one persistent kernel
    rnn_persistent_kernel<<<dim3(GRID_M, GRID_N), 128, 0, stream>>>(
        Wc, bias, act, out, ctr);

    // 3) loss
    loss_partial_kernel<<<LOSS_BLOCKS, 256, 0, stream>>>(out, labels, partial);
    loss_final_kernel<<<1, 256, 0, stream>>>(partial, out + OUT_ELEMS);
}
